// DarwinianRouter_72610717106441
// MI455X (gfx1250) — matrix operand order repaired and checked by reading, not compiled, not run on hardware
//
#include <hip/hip_runtime.h>
#include <math.h>

typedef __attribute__((ext_vector_type(16))) _Float16 v16h;
typedef __attribute__((ext_vector_type(4)))  _Float16 v4h;
typedef __attribute__((ext_vector_type(8)))  float    v8f;

#define DIM          4096
#define NEXP         64
#define TOPK         8
#define RSCALE       5.0f
#define TOK_PER_BLK  64
#define LDS_STRIDE   65   // 64 + 1 pad -> conflict-free column walks
#define KBODY        128  // 4 chunks of 32 fully unrolled per outer iteration
#define NCHUNK       (KBODY / 32)

__device__ __forceinline__ float softplus_f(float v) {
    // numerically stable log(1+exp(v)) == max(v,0) + log1p(exp(-|v|))
    return fmaxf(v, 0.0f) + log1pf(__expf(-fabsf(v)));
}

// One-shot fp32 -> f16 conversion of the signature matrix into workspace.
__global__ __launch_bounds__(256)
void sig_to_f16_kernel(const float* __restrict__ in, _Float16* __restrict__ out, int n4) {
    int i = blockIdx.x * blockDim.x + threadIdx.x;
    if (i < n4) {
        float4 v = ((const float4*)in)[i];
        v4h h;
        h.x = (_Float16)v.x; h.y = (_Float16)v.y;
        h.z = (_Float16)v.z; h.w = (_Float16)v.w;
        ((v4h*)out)[i] = h;
    }
}

template <bool USE_H16>
__global__ __launch_bounds__(256, 4)
void darwinian_router_kernel(const float*    __restrict__ x,     // [N, 4096]
                             const float*    __restrict__ sigF,  // [64, 4096] fp32 (fallback)
                             const _Float16* __restrict__ sigH,  // [64, 4096] f16 (preconverted)
                             float*          __restrict__ wts,   // [N, 8]
                             int*            __restrict__ idx,   // [N, 8]
                             float*          __restrict__ res)   // [N, 64]
{
    __shared__ float tile[TOK_PER_BLK * LDS_STRIDE];

    const int tid     = threadIdx.x;
    const int lane    = tid & 31;
    const int wave    = tid >> 5;
    const int mSub    = wave & 3;    // 16-token quarter of the 64-token block tile
    const int nPair   = wave >> 2;   // 32-expert half of the 64 experts
    const int tokBase = blockIdx.x * TOK_PER_BLK;

    const int m      = lane & 15;
    const int laneHi = lane >> 4;

    // A (16x32 f16): row = m; K elems off..off+15, off = laneHi*16
    const float* aPtr = x + (size_t)(tokBase + mSub * 16 + m) * DIM + laneHi * 16;
    // B (32x16 f16): col(expert) = m; K elems off..off+15, off = laneHi*16
    const size_t bOff0 = (size_t)(nPair * 32 + m) * DIM + laneHi * 16;
    const size_t bOff1 = bOff0 + (size_t)16 * DIM;
    const _Float16* bPtrH0 = sigH ? (sigH + bOff0) : nullptr;
    const _Float16* bPtrH1 = sigH ? (sigH + bOff1) : nullptr;
    const float*    bPtrF0 = sigF + bOff0;
    const float*    bPtrF1 = sigF + bOff1;

    v8f   c0 = {}, c1 = {};
    float sumsq = 0.0f;

    // Outer rolled loop; inner 4-chunk body fully unrolled and branch-free.
    // B fragments for the whole body are loaded up-front into distinct
    // registers so their (L2-hit) latency overlaps the A-side compute instead
    // of sitting on the critical path before each WMMA.
    for (int kb = 0; kb < DIM; kb += KBODY) {
        if (kb + 512 < DIM)   // pull the streaming-x K-stream ahead
            __builtin_prefetch(aPtr + kb + 512, 0, 0);

        v16h B0[NCHUNK], B1[NCHUNK];
        if constexpr (USE_H16) {
            #pragma unroll
            for (int j = 0; j < NCHUNK; ++j) {
                B0[j] = *(const v16h*)(bPtrH0 + kb + j * 32);  // 2x global_load_b128 each
                B1[j] = *(const v16h*)(bPtrH1 + kb + j * 32);
            }
        } else {
            #pragma unroll
            for (int j = 0; j < NCHUNK; ++j) {
                const float* p0 = bPtrF0 + kb + j * 32;
                const float* p1 = bPtrF1 + kb + j * 32;
                #pragma unroll
                for (int q = 0; q < 4; ++q) {
                    float4 t0 = *(const float4*)(p0 + 4 * q);
                    float4 t1 = *(const float4*)(p1 + 4 * q);
                    B0[j][4*q+0] = (_Float16)t0.x; B0[j][4*q+1] = (_Float16)t0.y;
                    B0[j][4*q+2] = (_Float16)t0.z; B0[j][4*q+3] = (_Float16)t0.w;
                    B1[j][4*q+0] = (_Float16)t1.x; B1[j][4*q+1] = (_Float16)t1.y;
                    B1[j][4*q+2] = (_Float16)t1.z; B1[j][4*q+3] = (_Float16)t1.w;
                }
            }
        }

        #pragma unroll
        for (int j = 0; j < NCHUNK; ++j) {
            const int kc = kb + j * 32;

            float4 a0 = *(const float4*)(aPtr + kc);
            float4 a1 = *(const float4*)(aPtr + kc + 4);
            float4 a2 = *(const float4*)(aPtr + kc + 8);
            float4 a3 = *(const float4*)(aPtr + kc + 12);

            // Branchless fused ||x||^2 partials; co-execute with XDL WMMA.
            sumsq = fmaf(a0.x, a0.x, sumsq); sumsq = fmaf(a0.y, a0.y, sumsq);
            sumsq = fmaf(a0.z, a0.z, sumsq); sumsq = fmaf(a0.w, a0.w, sumsq);
            sumsq = fmaf(a1.x, a1.x, sumsq); sumsq = fmaf(a1.y, a1.y, sumsq);
            sumsq = fmaf(a1.z, a1.z, sumsq); sumsq = fmaf(a1.w, a1.w, sumsq);
            sumsq = fmaf(a2.x, a2.x, sumsq); sumsq = fmaf(a2.y, a2.y, sumsq);
            sumsq = fmaf(a2.z, a2.z, sumsq); sumsq = fmaf(a2.w, a2.w, sumsq);
            sumsq = fmaf(a3.x, a3.x, sumsq); sumsq = fmaf(a3.y, a3.y, sumsq);
            sumsq = fmaf(a3.z, a3.z, sumsq); sumsq = fmaf(a3.w, a3.w, sumsq);

            v16h a;
            a[0]  = (_Float16)a0.x; a[1]  = (_Float16)a0.y; a[2]  = (_Float16)a0.z; a[3]  = (_Float16)a0.w;
            a[4]  = (_Float16)a1.x; a[5]  = (_Float16)a1.y; a[6]  = (_Float16)a1.z; a[7]  = (_Float16)a1.w;
            a[8]  = (_Float16)a2.x; a[9]  = (_Float16)a2.y; a[10] = (_Float16)a2.z; a[11] = (_Float16)a2.w;
            a[12] = (_Float16)a3.x; a[13] = (_Float16)a3.y; a[14] = (_Float16)a3.z; a[15] = (_Float16)a3.w;

            c0 = __builtin_amdgcn_wmma_f32_16x16x32_f16(false, a, false, B0[j], (short)0, c0, false, false);
            c1 = __builtin_amdgcn_wmma_f32_16x16x32_f16(false, a, false, B1[j], (short)0, c1, false, false);
        }
    }

    // Lanes m and m+16 hold the two K-halves of row m's sum of squares.
    float tot = sumsq + __shfl_xor(sumsq, 16, 32);
    float scl = RSCALE / fmaxf(sqrtf(tot), 1e-12f);

    // C/D layout: VGPR v, lanes 0-15 -> M=v, lanes 16-31 -> M=v+8; N = lane&15.
    #pragma unroll
    for (int v = 0; v < 8; ++v) {
        int   r = v + (laneHi << 3);
        float s = __shfl(scl, r, 32);           // row-r scale lives in lane r
        int   rr = mSub * 16 + r;
        tile[rr * LDS_STRIDE + nPair * 32 + m]      = c0[v] * s;
        tile[rr * LDS_STRIDE + nPair * 32 + 16 + m] = c1[v] * s;
    }
    __syncthreads();

    // Coalesced resonance writeback (global_store_b128)
    #pragma unroll
    for (int i4 = tid; i4 < TOK_PER_BLK * NEXP / 4; i4 += 256) {
        int t = i4 >> 4;
        int e = (i4 & 15) * 4;
        const float* tr = &tile[t * LDS_STRIDE + e];
        float4 o; o.x = tr[0]; o.y = tr[1]; o.z = tr[2]; o.w = tr[3];
        *(float4*)(res + (size_t)(tokBase + t) * NEXP + e) = o;
    }

    // Parallel top-8: 8 lanes per token, each lane owns 8 experts in registers.
    #pragma unroll
    for (int half = 0; half < TOK_PER_BLK / 32; ++half) {
        const int g     = (tid >> 3) + half * 32;  // token within block
        const int sub   = tid & 7;                 // lane within token group
        const int token = tokBase + g;
        const float* row = &tile[g * LDS_STRIDE + sub * 8];
        float l0 = row[0], l1 = row[1], l2 = row[2], l3 = row[3];
        float l4 = row[4], l5 = row[5], l6 = row[6], l7 = row[7];

        #pragma unroll
        for (int k = 0; k < TOPK; ++k) {
            // Local argmax: ascending scan with strict '>' keeps the lowest index.
            float bv = l0; int bl = 0;
            if (l1 > bv) { bv = l1; bl = 1; }
            if (l2 > bv) { bv = l2; bl = 2; }
            if (l3 > bv) { bv = l3; bl = 3; }
            if (l4 > bv) { bv = l4; bl = 4; }
            if (l5 > bv) { bv = l5; bl = 5; }
            if (l6 > bv) { bv = l6; bl = 6; }
            if (l7 > bv) { bv = l7; bl = 7; }
            int bi = sub * 8 + bl;

            // Cross-lane argmax over the 8-lane group (tie -> lowest expert index).
            #pragma unroll
            for (int off = 4; off > 0; off >>= 1) {
                float ov = __shfl_xor(bv, off, 8);
                int   oi = __shfl_xor(bi, off, 8);
                bool  take = (ov > bv) || ((ov == bv) && (oi < bi));
                bv = take ? ov : bv;
                bi = take ? oi : bi;
            }

            if (sub == 0) {
                wts[(size_t)token * TOPK + k] = softplus_f(bv);
                idx[(size_t)token * TOPK + k] = bi;
            }

            // Knock out the winner in its owner's registers.
            int lb = bi - sub * 8;
            l0 = (lb == 0) ? -INFINITY : l0;
            l1 = (lb == 1) ? -INFINITY : l1;
            l2 = (lb == 2) ? -INFINITY : l2;
            l3 = (lb == 3) ? -INFINITY : l3;
            l4 = (lb == 4) ? -INFINITY : l4;
            l5 = (lb == 5) ? -INFINITY : l5;
            l6 = (lb == 6) ? -INFINITY : l6;
            l7 = (lb == 7) ? -INFINITY : l7;
        }
    }
}

extern "C" void kernel_launch(void* const* d_in, const int* in_sizes, int n_in,
                              void* d_out, int out_size, void* d_ws, size_t ws_size,
                              hipStream_t stream) {
    const float* x   = (const float*)d_in[0];   // [N, 4096] fp32
    const float* sig = (const float*)d_in[1];   // [64, 4096] fp32 (unit rows)

    const int nTokens = in_sizes[0] / DIM;

    // Outputs concatenated flat in return order: weights f32, indices i32, resonance f32
    float* out = (float*)d_out;
    float* wts = out;
    int*   ids = (int*)(out + (size_t)nTokens * TOPK);
    float* res = out + (size_t)2 * nTokens * TOPK;

    const int    blocks   = nTokens / TOK_PER_BLK;
    const size_t sigBytes = (size_t)NEXP * DIM * 2;   // 512 KB f16 copy

    if (ws_size >= sigBytes) {
        _Float16* sigH = (_Float16*)d_ws;
        const int n4 = NEXP * DIM / 4;  // 65536 float4 groups
        sig_to_f16_kernel<<<n4 / 256, 256, 0, stream>>>(sig, sigH, n4);
        darwinian_router_kernel<true><<<blocks, 256, 0, stream>>>(x, sig, sigH, wts, ids, res);
    } else {
        darwinian_router_kernel<false><<<blocks, 256, 0, stream>>>(x, sig, nullptr, wts, ids, res);
    }
}
